// GraphDQN_55679956025498
// MI455X (gfx1250) — matrix operand order repaired and checked by reading, not compiled, not run on hardware
//
#include <hip/hip_runtime.h>
#include <hip/hip_bf16.h>

typedef __attribute__((ext_vector_type(16))) _Float16 v16h;
typedef __attribute__((ext_vector_type(8)))  float    v8f;

#define BN_EPS 1e-5f
#define NEG_SLOPE 0.01f

__device__ __forceinline__ float lrelu(float v) {
    return v >= 0.0f ? v : NEG_SLOPE * v;
}

// ---------------------------------------------------------------------------
// Degree / normalization
// ---------------------------------------------------------------------------
__global__ void deg_kernel(const int* __restrict__ dst, float* __restrict__ deg, int E) {
    int e = blockIdx.x * blockDim.x + threadIdx.x;
    if (e < E) atomicAdd(&deg[dst[e]], 1.0f);
}

__global__ void dinv_kernel(float* __restrict__ deg, int n) {
    int i = blockIdx.x * blockDim.x + threadIdx.x;
    if (i < n) deg[i] = __frsqrt_rn(deg[i] + 1.0f);   // +1 for self loop
}

// ---------------------------------------------------------------------------
// Pack W[K,F] (f32) into f16 WMMA-B lane layout:
//   Wp[((ct*KSTEPS + ks)*32 + lane)*16 + e] = W[k, ct*16 + (lane&15)]
//   with k = ks*32 + ((e>=8)?16:0) + ((lane>=16)?8:0) + (e&7)   (zero-padded for k >= K)
// ---------------------------------------------------------------------------
template<int K, int F>
__global__ void pack_w(const float* __restrict__ W, _Float16* __restrict__ Wp) {
    constexpr int KSTEPS = (K + 31) / 32;
    int tid = blockIdx.x * blockDim.x + threadIdx.x;
    if (tid >= (F / 16) * KSTEPS * 512) return;
    int e    = tid & 15;
    int lane = (tid >> 4) & 31;
    int rest = tid >> 9;                  // ct*KSTEPS + ks
    int ks = rest % KSTEPS;
    int ct = rest / KSTEPS;
    int col = ct * 16 + (lane & 15);
    int k   = ks * 32 + ((e >> 3) << 4) + ((lane >> 4) << 3) + (e & 7);
    Wp[tid] = (k < K) ? (_Float16)W[k * F + col] : (_Float16)0.0f;
}

// ---------------------------------------------------------------------------
// WMMA GEMM:  XW[nPad,F] = H[n,K] @ W[K,F]   (f16 inputs, f32 accumulate)
// One wave32 owns a 16-row strip: A fragments for the whole K live in
// registers; loop over the F/16 column tiles with one v16h B load per K-step.
// XW is padded to a multiple of 16 rows, so ALL stores are unguarded
// (pad-row writes land in scratch that nothing reads).
// ---------------------------------------------------------------------------
template<int K, int F>
__global__ void gemm_wmma(const float* __restrict__ H, const _Float16* __restrict__ Wp,
                          float* __restrict__ XW, int n) {
    constexpr int KSTEPS = (K + 31) / 32;
    int wave = blockIdx.x * (blockDim.x >> 5) + (threadIdx.x >> 5);
    int lane = threadIdx.x & 31;
    int rowTiles = (n + 15) >> 4;
    if (wave >= rowTiles) return;            // wave-uniform: EXEC stays all-1s
    int half = lane >> 4;                    // 0 or 1
    int l15  = lane & 15;
    int arow = wave * 16 + l15;
    int arowc = arow < n ? arow : n - 1;     // read clamp (input x is not padded)
    const float* hrow = H + (size_t)arowc * K;

    // Load + convert the A fragments for the entire K range.
    v16h a[KSTEPS];
#pragma unroll
    for (int ks = 0; ks < KSTEPS; ++ks) {
        int kb = ks * 32 + (half << 3);      // lower-half base: k0 + half*8
        float4 p0 = *(const float4*)(hrow + kb);
        float4 p1 = *(const float4*)(hrow + kb + 4);
        a[ks][0] = (_Float16)p0.x;  a[ks][1] = (_Float16)p0.y;
        a[ks][2] = (_Float16)p0.z;  a[ks][3] = (_Float16)p0.w;
        a[ks][4] = (_Float16)p1.x;  a[ks][5] = (_Float16)p1.y;
        a[ks][6] = (_Float16)p1.z;  a[ks][7] = (_Float16)p1.w;
        if constexpr (K % 32 == 0) {         // upper half: k0+16..k0+31
            float4 p2 = *(const float4*)(hrow + kb + 16);
            float4 p3 = *(const float4*)(hrow + kb + 20);
            a[ks][8]  = (_Float16)p2.x;  a[ks][9]  = (_Float16)p2.y;
            a[ks][10] = (_Float16)p2.z;  a[ks][11] = (_Float16)p2.w;
            a[ks][12] = (_Float16)p3.x;  a[ks][13] = (_Float16)p3.y;
            a[ks][14] = (_Float16)p3.z;  a[ks][15] = (_Float16)p3.w;
        } else {                             // K==16: zero-pad k>=16
            a[ks][8]  = (_Float16)0.0f;  a[ks][9]  = (_Float16)0.0f;
            a[ks][10] = (_Float16)0.0f;  a[ks][11] = (_Float16)0.0f;
            a[ks][12] = (_Float16)0.0f;  a[ks][13] = (_Float16)0.0f;
            a[ks][14] = (_Float16)0.0f;  a[ks][15] = (_Float16)0.0f;
        }
    }

    const v16h* B = (const v16h*)Wp;
    // Per-lane output base: row (wave*16 + half*8), col l15 (+ ct*16 below)
    float* outBase = XW + ((size_t)wave * 16 + (half << 3)) * F + l15;

#pragma unroll
    for (int ct = 0; ct < F / 16; ++ct) {
        v8f acc = {};
#pragma unroll
        for (int ks = 0; ks < KSTEPS; ++ks) {
            v16h b = B[(ct * KSTEPS + ks) * 32 + lane];
            acc = __builtin_amdgcn_wmma_f32_16x16x32_f16(false, a[ks], false, b,
                                                         (short)0, acc, false, false);
        }
        float* outp = outBase + ct * 16;
        // 8 unguarded stores at immediate offsets g*F*4 (row stride F)
#pragma unroll
        for (int g = 0; g < 8; ++g) outp[g * F] = acc[g];
    }
}

// ---------------------------------------------------------------------------
// Edge message passing: agg[dst] += dinv[src]*dinv[dst] * xw[src]
// One thread per (edge, 4-feature group). F = 1<<lgF.
// ---------------------------------------------------------------------------
__global__ void edge_scatter(const int* __restrict__ src, const int* __restrict__ dst,
                             const float* __restrict__ dinv,
                             const float* __restrict__ xw, float* __restrict__ agg,
                             int E, int lgF) {
    long long tid = (long long)blockIdx.x * blockDim.x + threadIdx.x;
    int lgfg = lgF - 2;                       // groups of 4 features
    long long total = (long long)E << lgfg;
    if (tid >= total) return;
    int e = (int)(tid >> lgfg);
    int f = ((int)tid & ((1 << lgfg) - 1)) << 2;
    int s = src[e], d = dst[e];
    float nrm = dinv[s] * dinv[d];
    const float4 v = *(const float4*)(xw + ((size_t)s << lgF) + f);
    float* out = agg + ((size_t)d << lgF) + f;
    atomicAdd(out + 0, nrm * v.x);
    atomicAdd(out + 1, nrm * v.y);
    atomicAdd(out + 2, nrm * v.z);
    atomicAdd(out + 3, nrm * v.w);
}

// ---------------------------------------------------------------------------
// Fused: self-loop + bias + eval BN + LeakyReLU
// ---------------------------------------------------------------------------
__global__ void post_kernel(const float* __restrict__ agg, const float* __restrict__ xw,
                            const float* __restrict__ dinv,
                            const float* __restrict__ b, const float* __restrict__ gamma,
                            const float* __restrict__ beta, const float* __restrict__ mean,
                            const float* __restrict__ var,
                            float* __restrict__ h, int n, int lgF) {
    int tid = blockIdx.x * blockDim.x + threadIdx.x;
    if (tid >= (n << lgF)) return;
    int i = tid >> lgF;
    int f = tid & ((1 << lgF) - 1);
    float di = dinv[i];
    float v = agg[tid] + di * di * xw[tid] + b[f];
    v = (v - mean[f]) * (gamma[f] * __frsqrt_rn(var[f] + BN_EPS)) + beta[f];
    h[tid] = lrelu(v);
}

// ---------------------------------------------------------------------------
// Global mean pool (atomic sums + counts)
// ---------------------------------------------------------------------------
__global__ void pool_kernel(const float* __restrict__ h, const int* __restrict__ batch,
                            float* __restrict__ pooled, int n) {
    int tid = blockIdx.x * blockDim.x + threadIdx.x;   // n * 32 threads
    if (tid >= n * 32) return;
    int i = tid >> 5;
    int f = (tid & 31) << 2;
    int g = batch[i];
    const float4 v = *(const float4*)(h + ((size_t)i << 7) + f);
    float* out = pooled + ((size_t)g << 7) + f;
    atomicAdd(out + 0, v.x);
    atomicAdd(out + 1, v.y);
    atomicAdd(out + 2, v.z);
    atomicAdd(out + 3, v.w);
}

__global__ void cnt_kernel(const int* __restrict__ batch, float* __restrict__ cnt, int n) {
    int i = blockIdx.x * blockDim.x + threadIdx.x;
    if (i < n) atomicAdd(&cnt[batch[i]], 1.0f);
}

// ---------------------------------------------------------------------------
// FC head (tiny, plain f32)
// ---------------------------------------------------------------------------
__global__ void fc_g(const float* __restrict__ pooled, const float* __restrict__ cnt,
                     const float* __restrict__ Wg, const float* __restrict__ bg,
                     float* __restrict__ g) {
    int gi = blockIdx.x, o = threadIdx.x;           // 128 x 128
    float c = cnt[gi];
    float inv = 1.0f / (c > 1.0f ? c : 1.0f);
    float s = bg[o];
    for (int k = 0; k < 128; ++k) s += (pooled[gi * 128 + k] * inv) * Wg[k * 128 + o];
    g[gi * 128 + o] = lrelu(s);
}

__global__ void fc_l(const float* __restrict__ low, const float* __restrict__ Wl,
                     const float* __restrict__ bl, float* __restrict__ l) {
    int gi = blockIdx.x, o = threadIdx.x;           // 128 x 128, K = 3
    float s = bl[o];
    for (int k = 0; k < 3; ++k) s += low[gi * 3 + k] * Wl[k * 128 + o];
    l[gi * 128 + o] = lrelu(s);
}

__global__ void fc_comb(const float* __restrict__ g, const float* __restrict__ l,
                        const float* __restrict__ Wc, const float* __restrict__ bc,
                        float* __restrict__ comb) {
    int gi = blockIdx.x, o = threadIdx.x;           // 128 x 128, K = 256 (concat)
    float s = bc[o];
    for (int k = 0; k < 128; ++k) {
        s += g[gi * 128 + k] * Wc[k * 128 + o];
        s += l[gi * 128 + k] * Wc[(128 + k) * 128 + o];
    }
    comb[gi * 128 + o] = lrelu(s);
}

__global__ void fc_out(const float* __restrict__ comb, const float* __restrict__ Wo,
                       const float* __restrict__ bo, float* __restrict__ out) {
    int gi = blockIdx.x, o = threadIdx.x;           // 128 x 4, K = 128
    float s = bo[o];
    for (int k = 0; k < 128; ++k) s += comb[gi * 128 + k] * Wo[k * 4 + o];
    out[gi * 4 + o] = s;
}

// ---------------------------------------------------------------------------
// Per-layer driver (compile-time shapes)
// ---------------------------------------------------------------------------
template<int K, int F>
static void run_gemm(const float* hin, const float* W, _Float16* Wp, float* xw,
                     int n, hipStream_t stream) {
    constexpr int KSTEPS = (K + 31) / 32;
    int packTot = (F / 16) * KSTEPS * 512;
    pack_w<K, F><<<(packTot + 255) / 256, 256, 0, stream>>>(W, Wp);
    int rowTiles = (n + 15) / 16;
    gemm_wmma<K, F><<<(rowTiles + 3) / 4, 128, 0, stream>>>(hin, Wp, xw, n);
}

// ---------------------------------------------------------------------------
// Launcher
// ---------------------------------------------------------------------------
extern "C" void kernel_launch(void* const* d_in, const int* in_sizes, int n_in,
                              void* d_out, int out_size, void* d_ws, size_t ws_size,
                              hipStream_t stream) {
    const float* x     = (const float*)d_in[0];     // [n,16]
    const float* low   = (const float*)d_in[1];     // [128,3]
    const int*   edge  = (const int*)d_in[34];      // [2,E] (row0=src, row1=dst)
    const int*   batch = (const int*)d_in[35];      // [n]
    const int n = in_sizes[0] / 16;
    const int E = in_sizes[34] / 2;
    const size_t nPad = (size_t)((n + 15) / 16) * 16;   // activations padded to 16 rows

    float* ws = (float*)d_ws;
    size_t off = 0;
    float* dinv   = ws + off; off += nPad;               // deg, then rsqrt in-place
    float* h      = ws + off; off += nPad * 128;
    float* xw     = ws + off; off += nPad * 128;
    float* agg    = ws + off; off += nPad * 128;
    float* pooled = ws + off; off += 128 * 128;
    float* cnt    = ws + off; off += 128;                // contiguous with pooled
    float* gbuf   = ws + off; off += 128 * 128;
    float* lbuf   = ws + off; off += 128 * 128;
    float* cbuf   = ws + off; off += 128 * 128;
    _Float16* Wp  = (_Float16*)(ws + off); off += 8192;  // 16K f16, 32B-aligned

    // D^{-1/2}
    hipMemsetAsync(dinv, 0, (size_t)n * sizeof(float), stream);
    deg_kernel<<<(E + 255) / 256, 256, 0, stream>>>(edge + E, dinv, E);
    dinv_kernel<<<(n + 255) / 256, 256, 0, stream>>>(dinv, n);

    static const int lgFs[4] = {5, 6, 7, 7};             // F = 32,64,128,128
    const float* hin = x;
    for (int L = 0; L < 4; ++L) {
        const int lgF = lgFs[L];
        const int F = 1 << lgF;
        const float* W    = (const float*)d_in[2 + L * 6];
        const float* b    = (const float*)d_in[3 + L * 6];
        const float* gam  = (const float*)d_in[4 + L * 6];
        const float* bet  = (const float*)d_in[5 + L * 6];
        const float* mean = (const float*)d_in[6 + L * 6];
        const float* var  = (const float*)d_in[7 + L * 6];

        switch (L) {
            case 0: run_gemm<16,  32 >(hin, W, Wp, xw, n, stream); break;
            case 1: run_gemm<32,  64 >(hin, W, Wp, xw, n, stream); break;
            case 2: run_gemm<64,  128>(hin, W, Wp, xw, n, stream); break;
            default: run_gemm<128, 128>(hin, W, Wp, xw, n, stream); break;
        }

        hipMemsetAsync(agg, 0, (size_t)n * F * sizeof(float), stream);
        long long total = (long long)E * (F / 4);
        edge_scatter<<<(unsigned)((total + 255) / 256), 256, 0, stream>>>(
            edge, edge + E, dinv, xw, agg, E, lgF);

        post_kernel<<<(n * F + 255) / 256, 256, 0, stream>>>(
            agg, xw, dinv, b, gam, bet, mean, var, h, n, lgF);
        hin = h;
    }

    // mean pool + head
    hipMemsetAsync(pooled, 0, (128 * 128 + 128) * sizeof(float), stream);
    pool_kernel<<<(n * 32 + 255) / 256, 256, 0, stream>>>(h, batch, pooled, n);
    cnt_kernel<<<(n + 255) / 256, 256, 0, stream>>>(batch, cnt, n);
    fc_g   <<<128, 128, 0, stream>>>(pooled, cnt, (const float*)d_in[26], (const float*)d_in[27], gbuf);
    fc_l   <<<128, 128, 0, stream>>>(low,         (const float*)d_in[28], (const float*)d_in[29], lbuf);
    fc_comb<<<128, 128, 0, stream>>>(gbuf, lbuf,  (const float*)d_in[30], (const float*)d_in[31], cbuf);
    fc_out <<<128,   4, 0, stream>>>(cbuf,        (const float*)d_in[32], (const float*)d_in[33], (float*)d_out);
}
